// GraphAttention_22866405884194
// MI455X (gfx1250) — compile-verified, matrix operand order checked by reading, not run on hardware
//
#include <hip/hip_runtime.h>

#define NN      50000
#define EE      1600000
#define IN_DIM  64
#define EIN_DIM 32
#define NH      4
#define DD      16
#define HD      64
#define ALPHA   0.2f

typedef _Float16 half_t;
typedef _Float16 v16h __attribute__((ext_vector_type(16)));
typedef float    v8f  __attribute__((ext_vector_type(8)));

// ---------------------------------------------------------------------------
// WMMA fragment layouts per CDNA5 ISA 7.12.2 (wave32).
// A (16x32 f16): lane holds row = lane&15; kb = (lane<16)?0:8.
//   VGPR pair p: K = kb+2p (p<4) or 16+kb+2(p-4) (p>=4); elements (K, K+1).
// B (32x16 f16): symmetric; lane holds col = colBase + (lane&15).
// C/D (16x16 f32): element v -> row = v + (lane<16?0:8), col = lane&15.
// ---------------------------------------------------------------------------
__device__ inline v16h load_frag_a_f32(const float* __restrict__ src, int lda) {
  int lane = threadIdx.x & 31;
  int row  = lane & 15;
  int kb   = (lane < 16) ? 0 : 8;
  v16h a;
#pragma unroll
  for (int p = 0; p < 8; ++p) {
    int K = (p < 4) ? (kb + 2 * p) : (16 + kb + 2 * (p - 4));
    float2 v = *(const float2*)(src + row * lda + K);
    a[2 * p]     = (half_t)v.x;
    a[2 * p + 1] = (half_t)v.y;
  }
  return a;
}

// One element of a pre-swizzled B fragment: inverse of the load mapping.
__device__ inline half_t frag_b_elem(const float* __restrict__ src, int ldb,
                                     int colBase, int lane, int elem) {
  int p  = elem >> 1, hi = elem & 1;
  int kb = (lane < 16) ? 0 : 8;
  int K  = ((p < 4) ? (kb + 2 * p) : (16 + kb + 2 * (p - 4))) + hi;
  return (half_t)src[K * ldb + colBase + (lane & 15)];
}

__device__ inline void atomicMaxF(float* addr, float val) {
  // IEEE ordering trick: signed-int max for non-negative, unsigned min for negative.
  if (val >= 0.0f) atomicMax((int*)addr, __float_as_int(val));
  else             atomicMin((unsigned int*)addr, __float_as_uint(val));
}

// ---------------------------------------------------------------------------
// K0: zero out/z, m = -FLT_MAX, and build WMMA-B-fragment-ready f16 copies of
// W (8 fragments: kchunk c in {0,1} x ntile j in {0..3}) and We (4 fragments).
// Layout: frag[lane][16 halves] -> each lane later does ONE aligned v16h load.
// ---------------------------------------------------------------------------
__global__ void init_kernel(float* __restrict__ out, float* __restrict__ m,
                            float* __restrict__ z, const float* __restrict__ W,
                            const float* __restrict__ We,
                            half_t* __restrict__ WF, half_t* __restrict__ WeF) {
  int i = blockIdx.x * blockDim.x + threadIdx.x;
  if (i < NN * HD) out[i] = 0.0f;
  if (i < NN * NH) { m[i] = -3.402823466e+38f; z[i] = 0.0f; }
  if (i < 8 * 32 * 16) {                  // W fragments: i = ((c*4+j)*32+lane)*16+elem
    int elem = i & 15, lane = (i >> 4) & 31, j = (i >> 9) & 3, c = i >> 11;
    WF[i] = frag_b_elem(W + c * 32 * HD, HD, j * 16, lane, elem);
  }
  if (i < 4 * 32 * 16) {                  // We fragments: i = (j*32+lane)*16+elem
    int elem = i & 15, lane = (i >> 4) & 31, j = i >> 9;
    WeF[i] = frag_b_elem(We, HD, j * 16, lane, elem);
  }
}

// ---------------------------------------------------------------------------
// K1: ft = x@W + b via WMMA f16. One wave per 16-node tile (3125 tiles).
// ---------------------------------------------------------------------------
__global__ void __launch_bounds__(256) node_transform(
    const float* __restrict__ x, const half_t* __restrict__ WF,
    const float* __restrict__ b, float* __restrict__ ft) {
  int wave = (blockIdx.x * blockDim.x + threadIdx.x) >> 5;
  if (wave >= NN / 16) return;                   // wave-uniform: EXEC stays all-ones
  int rowBase = wave * 16;
  int lane = threadIdx.x & 31;
  int nlo  = lane & 15;
  int rofs = (lane < 16) ? 0 : 8;

  v16h aLo = load_frag_a_f32(x + rowBase * IN_DIM, IN_DIM);       // K = 0..31
  v16h aHi = load_frag_a_f32(x + rowBase * IN_DIM + 32, IN_DIM);  // K = 32..63
  const v16h* WFv = (const v16h*)WF;

#pragma unroll
  for (int j = 0; j < 4; ++j) {
    float bias = b[j * 16 + nlo];
    v8f c = {bias, bias, bias, bias, bias, bias, bias, bias};
    v16h b0 = WFv[(0 * 4 + j) * 32 + lane];      // W rows 0..31, cols j*16..
    v16h b1 = WFv[(1 * 4 + j) * 32 + lane];      // W rows 32..63
    c = __builtin_amdgcn_wmma_f32_16x16x32_f16(false, aLo, false, b0, (short)0, c, false, false);
    c = __builtin_amdgcn_wmma_f32_16x16x32_f16(false, aHi, false, b1, (short)0, c, false, false);
    int col = j * 16 + nlo;
#pragma unroll
    for (int v = 0; v < 8; ++v)
      ft[(rowBase + v + rofs) * HD + col] = c[v];
  }
}

// ---------------------------------------------------------------------------
// K1b: a1/a2 per (node, head): 16-wide dot vs attn_l / attn_r.
// ---------------------------------------------------------------------------
__global__ void node_attn(const float* __restrict__ ft,
                          const float* __restrict__ attn_l,
                          const float* __restrict__ attn_r,
                          float* __restrict__ a1, float* __restrict__ a2) {
  int i = blockIdx.x * blockDim.x + threadIdx.x;
  if (i >= NN * NH) return;
  int node = i >> 2, h = i & 3;
  const float* f  = ft + node * HD + h * DD;
  const float* al = attn_l + h * DD;
  const float* ar = attn_r + h * DD;
  float s1 = 0.0f, s2 = 0.0f;
#pragma unroll
  for (int d = 0; d < DD; ++d) { float fv = f[d]; s1 += fv * al[d]; s2 += fv * ar[d]; }
  a1[i] = s1; a2[i] = s2;
}

// ---------------------------------------------------------------------------
// K2: edge logits + segment max. 256 edges/block; e staged via LDS (pad 33).
// ---------------------------------------------------------------------------
__global__ void __launch_bounds__(256) edge_logits(
    const float* __restrict__ e, const int* __restrict__ src, const int* __restrict__ dst,
    const float* __restrict__ attn_e, const float* __restrict__ a1,
    const float* __restrict__ a2, float* __restrict__ a, float* __restrict__ m) {
  __shared__ float es[256 * 33];
  __shared__ float s_ae[EIN_DIM * NH];
  int tid   = threadIdx.x;
  int ebase = blockIdx.x * 256;
  const float* eg = e + ebase * EIN_DIM;
  for (int i = tid; i < 256 * EIN_DIM; i += 256)
    es[(i >> 5) * 33 + (i & 31)] = eg[i];
  if (tid < EIN_DIM * NH) s_ae[tid] = attn_e[tid];
  __syncthreads();

  int eIdx = ebase + tid;
  float acc[NH] = {0.0f, 0.0f, 0.0f, 0.0f};
  const float* er = es + tid * 33;
#pragma unroll
  for (int k = 0; k < EIN_DIM; ++k) {
    float ek = er[k];
#pragma unroll
    for (int h = 0; h < NH; ++h) acc[h] += ek * s_ae[k * NH + h];
  }
  int s = src[eIdx], d = dst[eIdx];
  float4 v1 = *(const float4*)(a1 + s * NH);
  float4 v2 = *(const float4*)(a2 + d * NH);
  acc[0] += v1.x + v2.x; acc[1] += v1.y + v2.y;
  acc[2] += v1.z + v2.z; acc[3] += v1.w + v2.w;
#pragma unroll
  for (int h = 0; h < NH; ++h) acc[h] = (acc[h] > 0.0f) ? acc[h] : ALPHA * acc[h];
  *(float4*)(a + eIdx * NH) = make_float4(acc[0], acc[1], acc[2], acc[3]);
#pragma unroll
  for (int h = 0; h < NH; ++h) atomicMaxF(m + d * NH + h, acc[h]);
}

// ---------------------------------------------------------------------------
// K3: scores = exp(a - m[dst]) written in place over logits; z += scores.
// ---------------------------------------------------------------------------
__global__ void edge_scores(const int* __restrict__ dst, const float* __restrict__ m,
                            float* __restrict__ a, float* __restrict__ z) {
  int eIdx = blockIdx.x * blockDim.x + threadIdx.x;
  if (eIdx >= EE) return;
  int d = dst[eIdx];
  float4 av = *(const float4*)(a + eIdx * NH);
  float4 mv = *(const float4*)(m + d * NH);
  float4 sv = make_float4(__expf(av.x - mv.x), __expf(av.y - mv.y),
                          __expf(av.z - mv.z), __expf(av.w - mv.w));
  *(float4*)(a + eIdx * NH) = sv;
  atomicAdd(z + d * NH + 0, sv.x);
  atomicAdd(z + d * NH + 1, sv.y);
  atomicAdd(z + d * NH + 2, sv.z);
  atomicAdd(z + d * NH + 3, sv.w);
}

// ---------------------------------------------------------------------------
// K4: out[dst] += (scores/z[dst]) * ft[src] + (e@We + be), WMMA for eft.
// One wave per 16-edge tile (100000 tiles); D N-tile j == head j.
// ---------------------------------------------------------------------------
__global__ void __launch_bounds__(256) edge_aggregate(
    const float* __restrict__ e, const int* __restrict__ src, const int* __restrict__ dst,
    const half_t* __restrict__ WeF, const float* __restrict__ be,
    const float* __restrict__ ft, const float* __restrict__ scores,
    const float* __restrict__ z, float* __restrict__ out) {
  __shared__ int   s_src[8][16];
  __shared__ int   s_dst[8][16];
  __shared__ float s_w[8][16][NH];
  int tid  = threadIdx.x;
  int wblk = tid >> 5;
  int lane = tid & 31;
  int tile = blockIdx.x * 8 + wblk;
  if (tile >= EE / 16) return;                    // wave-uniform
  int ebase = tile * 16;

  if (lane < 16) {
    int eIdx = ebase + lane;
    int s = src[eIdx], d = dst[eIdx];
    s_src[wblk][lane] = s;
    s_dst[wblk][lane] = d;
    float4 sc = *(const float4*)(scores + eIdx * NH);
    float4 zv = *(const float4*)(z + d * NH);
    s_w[wblk][lane][0] = sc.x / zv.x;
    s_w[wblk][lane][1] = sc.y / zv.y;
    s_w[wblk][lane][2] = sc.z / zv.z;
    s_w[wblk][lane][3] = sc.w / zv.w;
  }
  __syncthreads();

  v16h af = load_frag_a_f32(e + ebase * EIN_DIM, EIN_DIM);
  const v16h* WeFv = (const v16h*)WeF;
  int nlo  = lane & 15;
  int rofs = (lane < 16) ? 0 : 8;

#pragma unroll
  for (int j = 0; j < NH; ++j) {
    float bias = be[j * 16 + nlo];
    v8f c = {bias, bias, bias, bias, bias, bias, bias, bias};
    v16h bf = WeFv[j * 32 + lane];                // one aligned 32B load
    c = __builtin_amdgcn_wmma_f32_16x16x32_f16(false, af, false, bf, (short)0, c, false, false);
    int col = j * 16 + nlo;
#pragma unroll
    for (int v = 0; v < 8; ++v) {
      int el = v + rofs;
      int sN = s_src[wblk][el];
      int dN = s_dst[wblk][el];
      float w = s_w[wblk][el][j];
      float val = w * ft[sN * HD + col] + c[v];
      atomicAdd(out + dN * HD + col, val);
    }
  }
}

// ---------------------------------------------------------------------------
extern "C" void kernel_launch(void* const* d_in, const int* in_sizes, int n_in,
                              void* d_out, int out_size, void* d_ws, size_t ws_size,
                              hipStream_t stream) {
  const float* x      = (const float*)d_in[0];
  const float* e      = (const float*)d_in[1];
  const int*   src    = (const int*)d_in[2];
  const int*   dst    = (const int*)d_in[3];
  const float* W      = (const float*)d_in[4];
  const float* b      = (const float*)d_in[5];
  const float* We     = (const float*)d_in[6];
  const float* be     = (const float*)d_in[7];
  const float* attn_l = (const float*)d_in[8];
  const float* attn_r = (const float*)d_in[9];
  const float* attn_e = (const float*)d_in[10];
  float* out = (float*)d_out;

  // Workspace layout (all f32 unless noted), ~41.6 MB total:
  float* ws = (float*)d_ws;
  float* ft = ws;                               // N*64   (12.8 MB, L2-resident)
  float* a1 = ft + (size_t)NN * HD;             // N*4
  float* a2 = a1 + (size_t)NN * NH;             // N*4
  float* m  = a2 + (size_t)NN * NH;             // N*4
  float* z  = m  + (size_t)NN * NH;             // N*4
  float* a  = z  + (size_t)NN * NH;             // E*4 (logits, reused as scores)
  half_t* WF  = (half_t*)(a + (size_t)EE * NH); // 8*32*16 f16 fragments (32B-aligned)
  half_t* WeF = WF + 8 * 32 * 16;               // 4*32*16 f16 fragments (32B-aligned)

  init_kernel<<<(NN * HD + 255) / 256, 256, 0, stream>>>(out, m, z, W, We, WF, WeF);
  node_transform<<<((NN / 16) * 32 + 255) / 256, 256, 0, stream>>>(x, WF, b, ft);
  node_attn<<<(NN * NH + 255) / 256, 256, 0, stream>>>(ft, attn_l, attn_r, a1, a2);
  edge_logits<<<EE / 256, 256, 0, stream>>>(e, src, dst, attn_e, a1, a2, a, m);
  edge_scores<<<(EE + 255) / 256, 256, 0, stream>>>(dst, m, a, z);
  edge_aggregate<<<EE / (16 * 8), 256, 0, stream>>>(e, src, dst, WeF, be, ft, a, z, out);
}
